// GWR_89988154785868
// MI455X (gfx1250) — matrix operand order repaired and checked by reading, not compiled, not run on hardware
//
#include <hip/hip_runtime.h>
#include <math.h>

// ---------------------------------------------------------------------------
// Fused VQ nearest-neighbor (top-2) for MI455X / gfx1250.
//
// dist^2(m,n) = c_m + [ w_n - 2 x.v ],  w_n = |v_n|^2 - 2e*Sum(v_n)
// GEMM via V_WMMA_F32_16X16X32_F16 with hi/lo f16 split of both operands
// (xh*vh + xh*vl + xl*vh, f32 accumulate => ~fp32 accuracy at f16 rate).
// Round-1 fixes: branchless top-2 (no EXEC branches in hot loop), and
// 3 independent WMMA accumulator chains per M-tile for latency hiding.
// ---------------------------------------------------------------------------

typedef __attribute__((ext_vector_type(16))) _Float16 v16h;
typedef __attribute__((ext_vector_type(8)))  float    v8f;

#define EPSQ   1e-6f
#define DIMK   128
#define NVEC   65536
#define NSAMP  4096
#define NTILES (NVEC / 16)     // 4096 N-tiles of 16
#define TM     2               // M-tiles (of 16 samples) per block
#define WAVES  8
#define BLOCK  (WAVES * 32)

// --- Pre-pass 1: pack V into WMMA-B-fragment-ordered f16 hi/lo arrays. ----
// B order chosen here (same k order as A): lane L holds column n = L&15; element j of the
// per-lane v16h holds K = c*32 + ((L>=16)?16:0) + j  (16 consecutive K).
__global__ __launch_bounds__(256) void pack_V(const float* __restrict__ V,
                                              _Float16* __restrict__ vh,
                                              _Float16* __restrict__ vl) {
    int idx = blockIdx.x * 256 + threadIdx.x;        // one per (nt, c, L)
    int L  = idx & 31;
    int c  = (idx >> 5) & 3;
    int nt = idx >> 7;
    if (nt >= NTILES) return;
    int n  = nt * 16 + (L & 15);
    int k0 = c * 32 + ((L & 16) ? 16 : 0);
    const float* src = V + (size_t)n * DIMK + k0;
    v16h h, l;
#pragma unroll
    for (int j = 0; j < 16; ++j) {
        float x = src[j];
        _Float16 hi = (_Float16)x;
        h[j] = hi;
        l[j] = (_Float16)(x - (float)hi);
    }
    size_t off = ((size_t)(nt * 4 + c) * 32 + L) * 16;
    *(v16h*)(vh + off) = h;
    *(v16h*)(vl + off) = l;
}

// --- Pre-pass 2: w[n] = |v_n|^2 - 2*eps*sum(v_n) ---------------------------
__global__ __launch_bounds__(256) void row_stats(const float* __restrict__ V,
                                                 float* __restrict__ w) {
    int n = blockIdx.x * 256 + threadIdx.x;
    if (n >= NVEC) return;
    const float* r = V + (size_t)n * DIMK;
    float sq = 0.f, sm = 0.f;
#pragma unroll 8
    for (int k = 0; k < DIMK; k += 4) {
        float4 v = *(const float4*)(r + k);
        sq += v.x * v.x + v.y * v.y + v.z * v.z + v.w * v.w;
        sm += v.x + v.y + v.z + v.w;
    }
    w[n] = sq - 2.0f * EPSQ * sm;
}

// --- Main fused kernel -----------------------------------------------------
__global__ __launch_bounds__(BLOCK) void vq_top2(const float*    __restrict__ S,
                                                 const _Float16* __restrict__ vh,
                                                 const _Float16* __restrict__ vl,
                                                 const float*    __restrict__ wn,
                                                 float*          __restrict__ out) {
    __shared__ __align__(32) _Float16 sA[TM][4][2][32][16];  // 16 KB
    __shared__ float  sc[TM * 16];
    __shared__ float4 cand[WAVES][32][TM][8];                // 64 KB

    const int mbase = blockIdx.x * (TM * 16);

    // Build A fragments (hi/lo).  A layout: lane L row m=L&15; element j:
    // K = c*32 + ((L&16)?16:0) + j.
    for (int e = threadIdx.x; e < TM * 4 * 32 * 16; e += BLOCK) {
        int j = e & 15;
        int L = (e >> 4) & 31;
        int c = (e >> 9) & 3;
        int t = e >> 11;
        int k = c * 32 + ((L & 16) ? 16 : 0) + j;
        int m = mbase + t * 16 + (L & 15);
        float x = S[(size_t)m * DIMK + k];
        _Float16 hi = (_Float16)x;
        sA[t][c][0][L][j] = hi;
        sA[t][c][1][L][j] = (_Float16)(x - (float)hi);
    }
    if (threadIdx.x < TM * 16) {
        const float* r = S + (size_t)(mbase + threadIdx.x) * DIMK;
        float sq = 0.f, sm = 0.f;
#pragma unroll 8
        for (int k = 0; k < DIMK; k += 4) {
            float4 v = *(const float4*)(r + k);
            sq += v.x * v.x + v.y * v.y + v.z * v.z + v.w * v.w;
            sm += v.x + v.y + v.z + v.w;
        }
        sc[threadIdx.x] = sq + 2.0f * EPSQ * sm + (float)DIMK * EPSQ * EPSQ;
    }
    __syncthreads();

    const int wv   = threadIdx.x >> 5;
    const int lane = threadIdx.x & 31;
    const int col  = lane & 15;

    float d1[TM][8], d2[TM][8];
    int   i1[TM][8], i2[TM][8];
#pragma unroll
    for (int t = 0; t < TM; ++t)
#pragma unroll
        for (int r = 0; r < 8; ++r) {
            d1[t][r] = INFINITY; d2[t][r] = INFINITY;
            i1[t][r] = 0x7fffffff; i2[t][r] = 0x7fffffff;
        }

    const int nt0 = wv * (NTILES / WAVES);
    const int nt1 = nt0 + (NTILES / WAVES);
    for (int nt = nt0; nt < nt1; ++nt) {
        v16h Bh[4], Bl[4];
#pragma unroll
        for (int c = 0; c < 4; ++c) {
            size_t off = ((size_t)(nt * 4 + c) * 32 + lane) * 16;
            Bh[c] = *(const v16h*)(vh + off);
            Bl[c] = *(const v16h*)(vl + off);
        }
        const float wcol = wn[nt * 16 + col];
        const int   ncol = nt * 16 + col;

#pragma unroll
        for (int t = 0; t < TM; ++t) {
            // Three independent accumulator chains (ILP across WMMA latency).
            v8f ahh = {}, ahl = {}, alh = {};
#pragma unroll
            for (int c = 0; c < 4; ++c) {
                v16h Ah = *(const v16h*)&sA[t][c][0][lane][0];
                v16h Al = *(const v16h*)&sA[t][c][1][lane][0];
                ahh = __builtin_amdgcn_wmma_f32_16x16x32_f16(false, Ah, false, Bh[c], (short)0, ahh, false, false);
                ahl = __builtin_amdgcn_wmma_f32_16x16x32_f16(false, Ah, false, Bl[c], (short)0, ahl, false, false);
                alh = __builtin_amdgcn_wmma_f32_16x16x32_f16(false, Al, false, Bh[c], (short)0, alh, false, false);
            }
            // key = w_n - 2*(hh+hl+lh), folded as 3 FMAs; branchless top-2.
#pragma unroll
            for (int r = 0; r < 8; ++r) {
                float key = fmaf(-2.0f, ahh[r],
                            fmaf(-2.0f, ahl[r],
                            fmaf(-2.0f, alh[r], wcol)));
                const float od1 = d1[t][r], od2 = d2[t][r];
                const int   oi1 = i1[t][r], oi2 = i2[t][r];
                const bool  lt1 = key < od1;
                const bool  lt2 = key < od2;
                d2[t][r] = lt1 ? od1 : (lt2 ? key  : od2);
                i2[t][r] = lt1 ? oi1 : (lt2 ? ncol : oi2);
                d1[t][r] = lt1 ? key  : od1;
                i1[t][r] = lt1 ? ncol : oi1;
            }
        }
    }

#pragma unroll
    for (int t = 0; t < TM; ++t)
#pragma unroll
        for (int r = 0; r < 8; ++r)
            cand[wv][lane][t][r] = make_float4(d1[t][r], __int_as_float(i1[t][r]),
                                               d2[t][r], __int_as_float(i2[t][r]));
    __syncthreads();

    if (threadIdx.x < TM * 16) {
        const int t  = threadIdx.x >> 4;
        const int m  = threadIdx.x & 15;
        const int lb = (m & 8) ? 16 : 0;
        const int r  = m & 7;
        float D1 = INFINITY, D2 = INFINITY;
        int   I1 = 0x7fffffff, I2 = 0x7fffffff;
        auto ins = [&](float d, int i) {
            if (d < D1 || (d == D1 && i < I1)) { D2 = D1; I2 = I1; D1 = d; I1 = i; }
            else if (d < D2 || (d == D2 && i < I2)) { D2 = d; I2 = i; }
        };
        for (int w8 = 0; w8 < WAVES; ++w8)
            for (int j = 0; j < 16; ++j) {
                float4 c4 = cand[w8][lb + j][t][r];
                ins(c4.x, __float_as_int(c4.y));
                ins(c4.z, __float_as_int(c4.w));
            }
        float cm = sc[threadIdx.x];
        float dd = sqrtf(fmaxf(cm + D1, 0.0f));
        int gm = mbase + threadIdx.x;
        out[gm]             = (float)I1;        // b
        out[NSAMP + gm]     = (float)I2;        // s
        out[2 * NSAMP + gm] = expf(-dd);        // a
    }
}

extern "C" void kernel_launch(void* const* d_in, const int* in_sizes, int n_in,
                              void* d_out, int out_size, void* d_ws, size_t ws_size,
                              hipStream_t stream) {
    const float* S = (const float*)d_in[0];   // samples 4096x128 f32
    const float* V = (const float*)d_in[1];   // V 65536x128 f32

    _Float16* vh = (_Float16*)d_ws;
    _Float16* vl = vh + (size_t)NVEC * DIMK;
    float*    w  = (float*)(vl + (size_t)NVEC * DIMK);

    pack_V   <<<(NTILES * 4 * 32) / 256, 256, 0, stream>>>(V, vh, vl);
    row_stats<<<NVEC / 256,            256, 0, stream>>>(V, w);
    vq_top2  <<<NSAMP / (TM * 16),   BLOCK, 0, stream>>>(S, vh, vl, w, (float*)d_out);
}
